// Attention_67697274519924
// MI455X (gfx1250) — matrix operand order repaired and checked by reading, not compiled, not run on hardware
//
#include <hip/hip_runtime.h>
#include <hip/hip_bf16.h>

// PVT spatial-reduction attention for MI455X (gfx1250).
// HBM-bound on relative_pos (128 MB); flash-style online softmax so the
// [4,8,4096,1024] score matrix is never materialized. All GEMMs via
// v_wmma_f32_16x16x32_f16 (f32 accumulate), head-dim padded 48 -> 64.
// Softmax runs in base-2 domain (log2e folded into the Q pre-scale) so
// exponentials map 1:1 onto v_exp_f32.

typedef __attribute__((ext_vector_type(16))) _Float16 v16h;
typedef __attribute__((ext_vector_type(8)))  float    v8f;

#define DIM   384
#define HEADS 8
#define NQ    4096   // 64*64
#define NK    1024   // 32*32
#define HD    48
#define HDP   64     // padded head dim (multiple of 32 for WMMA K)
#define BATCH 4
#define LOG2E 1.4426950408889634f

// ---- WMMA fragment index helpers (per CDNA5 ISA 7.12.2, wave32) ----
// A (16-bit, 16xK): lane holds row = lane&15; element e -> k:
__device__ __forceinline__ int kA(int e, int grp) {
  return (e & 7) + ((e >> 3) << 4) + (grp << 3);
}
// B (16-bit, 32x16 chunk): lane holds col = lane&15; element e -> k:
__device__ __forceinline__ int kB(int e, int grp) {
  return (e & 7) + ((e >> 3) << 4) + (grp << 3);
}
// C/D f32: vgpr r holds row = r + 8*grp, col = lane&15.

__device__ __forceinline__ v8f wmma_f16(v16h a, v16h b, v8f c) {
  return __builtin_amdgcn_wmma_f32_16x16x32_f16(
      false, a, false, b, (short)0, c, false, false);
}

// =====================================================================
// 0) zero only the pad columns (d = 48..63) of Q/K/V (ws is poisoned)
// =====================================================================
__global__ void fill_pads(_Float16* __restrict__ Qh, _Float16* __restrict__ Kh,
                          _Float16* __restrict__ Vt) {
  const unsigned qrows = BATCH * HEADS * NQ;       // rows with 16 pad halves
  const unsigned krows = BATCH * HEADS * NK;
  const unsigned qk = (qrows + krows) * 16;
  const unsigned vpad = BATCH * HEADS * 16 * NK;   // Vt pad rows are contiguous
  unsigned i = blockIdx.x * blockDim.x + threadIdx.x;
  if (i < qk) {
    unsigned e = i & 15, row = i >> 4;
    if (row < qrows) Qh[(size_t)row * HDP + HD + e] = (_Float16)0.f;
    else             Kh[(size_t)(row - qrows) * HDP + HD + e] = (_Float16)0.f;
  } else if (i < qk + vpad) {
    unsigned j = i - qk;
    unsigned bh = j / (16 * NK), rem = j % (16 * NK);
    Vt[(size_t)bh * HDP * NK + HD * NK + rem] = (_Float16)0.f;
  }
}

// =====================================================================
// 1) depthwise 2x2 stride-2 conv + BatchNorm(eval) -> xk [B, NK, C] f32
// =====================================================================
__global__ void sr_bn_kernel(const float* __restrict__ x,
                             const float* __restrict__ sr_w,
                             const float* __restrict__ sr_b,
                             const float* __restrict__ gamma,
                             const float* __restrict__ beta,
                             const float* __restrict__ mean,
                             const float* __restrict__ var,
                             float* __restrict__ xk) {
  unsigned i = blockIdx.x * blockDim.x + threadIdx.x;
  const unsigned total = BATCH * NK * DIM;
  if (i >= total) return;
  int c  = i % DIM;
  int nk = (i / DIM) % NK;
  int b  = i / (DIM * NK);
  int oh = nk >> 5, ow = nk & 31;          // 32x32 output grid
  int h0 = oh * 2,  w0 = ow * 2;           // 64x64 input grid
  const float* xb = x + ((size_t)b * NQ) * DIM + c;
  float s = 0.f;
#pragma unroll
  for (int dy = 0; dy < 2; ++dy)
#pragma unroll
    for (int dx = 0; dx < 2; ++dx)
      s += xb[(size_t)((h0 + dy) * 64 + (w0 + dx)) * DIM] * sr_w[c * 4 + dy * 2 + dx];
  s += sr_b[c];
  float inv = rsqrtf(var[c] + 1e-5f) * gamma[c];
  xk[i] = (s - mean[c]) * inv + beta[c];
}

// =====================================================================
// 2) generic WMMA GEMM: out = A[M,K] @ W[NC,K]^T, with epilogue modes
// =====================================================================
enum { MODE_Q = 0, MODE_K = 1, MODE_V = 2, MODE_PROJ = 3 };

template <typename AT, int MODE>
__global__ __launch_bounds__(128)
void gemm_wmma_kernel(const AT* __restrict__ A, const float* __restrict__ W,
                      const float* __restrict__ bias, void* __restrict__ outp,
                      int M, int K, int NC, float scale) {
  const int lane = threadIdx.x & 31;
  const int wv   = threadIdx.x >> 5;
  const int grp  = lane >> 4;
  const int ln   = lane & 15;
  const int m0   = blockIdx.x * 64 + wv * 16;   // 16 rows per wave
  const int n0   = blockIdx.y * 64;             // 4 col-tiles per wave

  v8f acc[4] = {};
  for (int k0 = 0; k0 < K; k0 += 32) {
    v16h a;
#pragma unroll
    for (int e = 0; e < 16; ++e)
      a[e] = (_Float16)(float)A[(size_t)(m0 + ln) * K + k0 + kA(e, grp)];
#pragma unroll
    for (int t = 0; t < 4; ++t) {
      v16h bfr;
#pragma unroll
      for (int e = 0; e < 16; ++e)
        bfr[e] = (_Float16)W[(size_t)(n0 + t * 16 + ln) * K + k0 + kB(e, grp)];
      acc[t] = wmma_f16(a, bfr, acc[t]);
    }
  }

#pragma unroll
  for (int t = 0; t < 4; ++t) {
#pragma unroll
    for (int r = 0; r < 8; ++r) {
      int m = m0 + r + 8 * grp;
      int n = n0 + t * 16 + ln;
      float v = acc[t][r];
      if (MODE == MODE_PROJ) {
        ((float*)outp)[(size_t)m * NC + n] = v + bias[n];
      } else {
        int h = n / HD, d = n % HD;
        _Float16 hv = (_Float16)(v * scale);
        if (MODE == MODE_Q) {
          int b = m >> 12, row = m & (NQ - 1);
          ((_Float16*)outp)[(((size_t)(b * HEADS + h) * NQ) + row) * HDP + d] = hv;
        } else if (MODE == MODE_K) {
          int b = m >> 10, row = m & (NK - 1);
          ((_Float16*)outp)[(((size_t)(b * HEADS + h) * NK) + row) * HDP + d] = hv;
        } else {  // MODE_V: transposed store Vt[bh][d][row]
          int b = m >> 10, row = m & (NK - 1);
          ((_Float16*)outp)[(((size_t)(b * HEADS + h) * HDP) + d) * NK + row] = hv;
        }
      }
    }
  }
}

// =====================================================================
// 3) flash attention: S = Q K^T (+rel_pos), online softmax (base-2),
//    O = P V.  4 waves/block; wave owns 16 query rows; Nk chunks of 32.
// =====================================================================
__global__ __launch_bounds__(128)
void attn_kernel(const _Float16* __restrict__ Qh, const _Float16* __restrict__ Kh,
                 const _Float16* __restrict__ Vt, const float* __restrict__ rel,
                 _Float16* __restrict__ Oh) {
  __shared__ _Float16 plds[4][16 * 32];   // per-wave P staging (D-layout -> A-layout)
  const int lane = threadIdx.x & 31;
  const int wv   = threadIdx.x >> 5;
  const int grp  = lane >> 4;
  const int ln   = lane & 15;

  const int tilesPerBH = NQ / 64;
  const int bh   = blockIdx.x / tilesPerBH;     // b*HEADS + h
  const int tile = blockIdx.x % tilesPerBH;
  const int h    = bh & (HEADS - 1);
  const int b    = bh >> 3;
  const int m0   = tile * 64 + wv * 16;

  // Q fragments for k = 0..63 (two 16x32 A-fragments); Q carries
  // hd^-0.5 * log2e, so S is already in the base-2 domain.
  v16h qa[2];
#pragma unroll
  for (int ks = 0; ks < 2; ++ks)
#pragma unroll
    for (int e = 0; e < 16; ++e)
      qa[ks][e] = Qh[((size_t)bh * NQ + m0 + ln) * HDP + 32 * ks + kA(e, grp)];

  float mrow[8], lrow[8];
  v8f Oacc[4] = {};
#pragma unroll
  for (int r = 0; r < 8; ++r) { mrow[r] = -1e30f; lrow[r] = 0.f; }

  const float* relbase = rel + (size_t)h * NQ * NK;
  const _Float16* Kb = Kh + (size_t)bh * NK * HDP;
  const _Float16* Vb = Vt + (size_t)bh * HDP * NK;
  _Float16* pw = plds[wv];

  for (int j0 = 0; j0 < NK; j0 += 32) {
    if (j0 + 32 < NK)  // prefetch next rel_pos chunk (global_prefetch_b8)
      __builtin_prefetch(&relbase[(size_t)(m0 + ln) * NK + j0 + 32], 0, 1);

    // ---- S = Q K^T + rel * log2e  (base-2 domain) ----
    v8f s[2];
#pragma unroll
    for (int nt = 0; nt < 2; ++nt) {
      v8f accS = {};
#pragma unroll
      for (int ks = 0; ks < 2; ++ks) {
        v16h kb;
#pragma unroll
        for (int e = 0; e < 16; ++e)
          kb[e] = Kb[(size_t)(j0 + nt * 16 + ln) * HDP + 32 * ks + kB(e, grp)];
        accS = wmma_f16(qa[ks], kb, accS);
      }
#pragma unroll
      for (int r = 0; r < 8; ++r)
        accS[r] = fmaf(relbase[(size_t)(m0 + r + 8 * grp) * NK + j0 + nt * 16 + ln],
                       LOG2E, accS[r]);
      s[nt] = accS;
    }

    // ---- online softmax (rows live in 16-lane halves; exp2 == v_exp_f32) ----
    float pl[2][8];
#pragma unroll
    for (int r = 0; r < 8; ++r) {
      float mx = fmaxf(s[0][r], s[1][r]);
#pragma unroll
      for (int off = 1; off < 16; off <<= 1)
        mx = fmaxf(mx, __shfl_xor(mx, off, 16));
      float mn = fmaxf(mrow[r], mx);
      float alpha = exp2f(mrow[r] - mn);
      mrow[r] = mn;
      float p0 = exp2f(s[0][r] - mn);
      float p1 = exp2f(s[1][r] - mn);
      float ps = p0 + p1;
#pragma unroll
      for (int off = 1; off < 16; off <<= 1)
        ps += __shfl_xor(ps, off, 16);
      lrow[r] = lrow[r] * alpha + ps;
#pragma unroll
      for (int t = 0; t < 4; ++t) Oacc[t][r] *= alpha;
      pl[0][r] = p0; pl[1][r] = p1;
    }

    // ---- P: D-layout -> LDS -> A-fragment ----
#pragma unroll
    for (int nt = 0; nt < 2; ++nt)
#pragma unroll
      for (int r = 0; r < 8; ++r)
        pw[(r + 8 * grp) * 32 + nt * 16 + ln] = (_Float16)pl[nt][r];
    __builtin_amdgcn_wave_barrier();
    asm volatile("s_wait_dscnt 0" ::: "memory");
    v16h pa;
#pragma unroll
    for (int e = 0; e < 16; ++e)
      pa[e] = pw[ln * 32 + kA(e, grp)];
    __builtin_amdgcn_wave_barrier();

    // ---- O += P V ----
#pragma unroll
    for (int t = 0; t < 4; ++t) {
      v16h vb;
#pragma unroll
      for (int e = 0; e < 16; ++e)
        vb[e] = Vb[(size_t)(t * 16 + ln) * NK + j0 + kB(e, grp)];
      Oacc[t] = wmma_f16(pa, vb, Oacc[t]);
    }
  }

  // ---- finalize: O /= l, drop pad cols, write [B, NQ, 384] f16 ----
#pragma unroll
  for (int t = 0; t < 4; ++t) {
#pragma unroll
    for (int r = 0; r < 8; ++r) {
      int d = t * 16 + ln;
      if (d < HD) {
        int m = m0 + r + 8 * grp;
        Oh[((size_t)b * NQ + m) * DIM + h * HD + d] =
            (_Float16)(Oacc[t][r] / lrow[r]);
      }
    }
  }
}

// =====================================================================
// launch
// =====================================================================
extern "C" void kernel_launch(void* const* d_in, const int* in_sizes, int n_in,
                              void* d_out, int out_size, void* d_ws, size_t ws_size,
                              hipStream_t stream) {
  const float* x    = (const float*)d_in[0];
  const float* rel  = (const float*)d_in[1];
  const float* Wq   = (const float*)d_in[4];
  const float* Wk   = (const float*)d_in[5];
  const float* Wv   = (const float*)d_in[6];
  const float* Wp   = (const float*)d_in[7];
  const float* bp   = (const float*)d_in[8];
  const float* sr_w = (const float*)d_in[9];
  const float* sr_b = (const float*)d_in[10];
  const float* gm   = (const float*)d_in[11];
  const float* bt   = (const float*)d_in[12];
  const float* mu   = (const float*)d_in[13];
  const float* vr   = (const float*)d_in[14];
  float* out = (float*)d_out;

  // workspace layout
  char* ws = (char*)d_ws;
  float*     xk = (float*)ws;                                       // 6,291,456 B
  _Float16*  Qh = (_Float16*)(ws + 6291456);                        // 16,777,216 B
  _Float16*  Kh = (_Float16*)(ws + 6291456 + 16777216);             //  4,194,304 B
  _Float16*  Vt = (_Float16*)(ws + 6291456 + 16777216 + 4194304);   //  4,194,304 B
  _Float16*  Oh = (_Float16*)(ws + 6291456 + 16777216 + 2 * 4194304); // 12,582,912 B

  const float qscale = 0.14433756729740643f * LOG2E;  // 48^-0.5 * log2e

  // zero only pad columns of Q/K/V (poisoned ws)
  {
    unsigned npad = (BATCH * HEADS * NQ + BATCH * HEADS * NK) * 16u +
                    BATCH * HEADS * 16u * NK;
    fill_pads<<<(npad + 255) / 256, 256, 0, stream>>>(Qh, Kh, Vt);
  }

  // spatial reduction (depthwise conv + BN)
  {
    unsigned tot = BATCH * NK * DIM;
    sr_bn_kernel<<<(tot + 255) / 256, 256, 0, stream>>>(x, sr_w, sr_b, gm, bt, mu, vr, xk);
  }

  // projections
  gemm_wmma_kernel<float, MODE_Q><<<dim3((BATCH * NQ) / 64, DIM / 64), 128, 0, stream>>>(
      x, Wq, nullptr, Qh, BATCH * NQ, DIM, DIM, qscale);
  gemm_wmma_kernel<float, MODE_K><<<dim3((BATCH * NK) / 64, DIM / 64), 128, 0, stream>>>(
      xk, Wk, nullptr, Kh, BATCH * NK, DIM, DIM, 1.f);
  gemm_wmma_kernel<float, MODE_V><<<dim3((BATCH * NK) / 64, DIM / 64), 128, 0, stream>>>(
      xk, Wv, nullptr, Vt, BATCH * NK, DIM, DIM, 1.f);

  // flash attention
  attn_kernel<<<BATCH * HEADS * (NQ / 64), 128, 0, stream>>>(Qh, Kh, Vt, rel, Oh);

  // output projection (f32 out, + bias)
  gemm_wmma_kernel<_Float16, MODE_PROJ><<<dim3((BATCH * NQ) / 64, DIM / 64), 128, 0, stream>>>(
      Oh, Wp, bp, out, BATCH * NQ, DIM, DIM, 1.f);
}
